// ModulatedConv2d_80796924772696
// MI455X (gfx1250) — matrix operand order repaired and checked by reading, not compiled, not run on hardware
//
#include <hip/hip_runtime.h>
#include <hip/hip_bf16.h>

// ---------------------------------------------------------------------------
// ModulatedConv2d (StyleGAN2) for gfx1250 / MI455X
//   x:(8,512,64,64) f32, style:(8,512), weight:(1,512,512,3,3),
//   mod_weight:(512,512), mod_bias:(512,)  ->  out:(8,512,64,64) f32
// Strategy: 9 shifted GEMMs (per kernel tap) in f16 WMMA with f32 accum.
// Staging uses gfx1250 GLOBAL_LOAD_ASYNC_TO_LDS_B128 (ASYNCcnt) to move
// L2-resident packed operands into double-buffered LDS without VGPR traffic.
// ---------------------------------------------------------------------------

typedef _Float16 v16h __attribute__((ext_vector_type(16)));
typedef _Float16 v8h  __attribute__((ext_vector_type(8)));
typedef float    v8f  __attribute__((ext_vector_type(8)));

union Frag { v16h v; v8h h[2]; };

#define IN_CH   512
#define OUT_CH  512
#define HW      64
#define NPIX    4096          // 64*64
#define KTAPS   9
#define BATCH   8
#define CONV_SCALE 0.014731391274719739f   // 1/sqrt(512*9)
#define LIN_SCALE  0.044194173824159216f   // 1/sqrt(512)

// ---------------------------------------------------------------------------
// Kernel 1: s[b][ic] = style[b] . (mod_weight[ic] * lin_scale) + mod_bias[ic]
// ---------------------------------------------------------------------------
__global__ __launch_bounds__(256) void k_modstyle(
    const float* __restrict__ style, const float* __restrict__ mw,
    const float* __restrict__ mb, float* __restrict__ s)
{
    int idx = blockIdx.x * 256 + threadIdx.x;      // 0..4095
    int b  = idx >> 9;
    int ic = idx & 511;
    float acc = 0.f;
    const float* st = style + b * IN_CH;
    const float* wr = mw + (size_t)ic * IN_CH;
    for (int j = 0; j < IN_CH; ++j) acc += st[j] * wr[j];
    s[idx] = acc * LIN_SCALE + mb[ic];             // LR_MUL == 1
}

// ---------------------------------------------------------------------------
// Kernel 2: demod[b][oc] = rsqrt( sum_{ic,kk} (conv_scale*W*s)^2 + eps )
// One 256-thread block per (b,oc).
// ---------------------------------------------------------------------------
__global__ __launch_bounds__(256) void k_demod(
    const float* __restrict__ weight, const float* __restrict__ s,
    float* __restrict__ demod)
{
    int oc = blockIdx.x;
    int b  = blockIdx.y;
    const float* wrow = weight + (size_t)oc * (IN_CH * KTAPS);
    const float* srow = s + b * IN_CH;
    float sum = 0.f;
    for (int j = threadIdx.x; j < IN_CH * KTAPS; j += 256) {
        int ic = j / KTAPS;
        float v = CONV_SCALE * wrow[j] * srow[ic];
        sum += v * v;
    }
    __shared__ float red[256];
    red[threadIdx.x] = sum;
    __syncthreads();
    for (int off = 128; off > 0; off >>= 1) {
        if (threadIdx.x < off) red[threadIdx.x] += red[threadIdx.x + off];
        __syncthreads();
    }
    if (threadIdx.x == 0) demod[b * OUT_CH + oc] = rsqrtf(red[0] + 1e-8f);
}

// ---------------------------------------------------------------------------
// Kernel 3: pack f16 weights Wp[((b*9+kk)*512 + oc)*512 + ic]
//           = conv_scale * W[oc][ic][kk] * s[b][ic] * demod[b][oc]
// ---------------------------------------------------------------------------
__global__ __launch_bounds__(256) void k_packw(
    const float* __restrict__ weight, const float* __restrict__ s,
    const float* __restrict__ demod, _Float16* __restrict__ Wp)
{
    size_t idx = (size_t)blockIdx.x * 256 + threadIdx.x;   // 18,874,368 total
    int ic = (int)(idx & 511);
    size_t t = idx >> 9;
    int oc = (int)(t & 511);
    t >>= 9;
    int kk = (int)(t % KTAPS);
    int b  = (int)(t / KTAPS);
    float v = CONV_SCALE * weight[(size_t)oc * (IN_CH * KTAPS) + (size_t)ic * KTAPS + kk]
            * s[b * IN_CH + ic] * demod[b * OUT_CH + oc];
    Wp[idx] = (_Float16)v;
}

// ---------------------------------------------------------------------------
// Kernel 4: transpose x (b,ic,p) -> Xt (b,p,ic) in f16, 32x32 LDS tiles
// grid (NPIX/32, IN_CH/32, BATCH), block (32,8)
// ---------------------------------------------------------------------------
__global__ __launch_bounds__(256) void k_packx(
    const float* __restrict__ x, _Float16* __restrict__ Xt)
{
    __shared__ float tile[32][33];
    int b  = blockIdx.z;
    int p0 = blockIdx.x * 32;
    int c0 = blockIdx.y * 32;
    int tx = threadIdx.x;       // 0..31
    int ty = threadIdx.y;       // 0..7
    for (int i = ty; i < 32; i += 8)
        tile[i][tx] = x[((size_t)b * IN_CH + c0 + i) * NPIX + p0 + tx];
    __syncthreads();
    for (int i = ty; i < 32; i += 8)
        Xt[((size_t)b * NPIX + p0 + i) * IN_CH + c0 + tx] = (_Float16)tile[tx][i];
}

// ---------------------------------------------------------------------------
// Kernel 5: main WMMA GEMM.
// Block tile: 128 oc x 128 px, one batch. 8 waves: 4 oc-groups x 2 px-groups,
// each wave -> 2x4 = 8 accumulators of v_wmma_f32_16x16x32_f16.
// K loop: kk (9 taps) x ic chunks of 32 -> 144 steps, double-buffered LDS
// filled by gfx1250 async global->LDS copies (ASYNCcnt).
// ---------------------------------------------------------------------------
__global__ __launch_bounds__(256) void k_mconv(
    const _Float16* __restrict__ Wp, const _Float16* __restrict__ Xt,
    float* __restrict__ out)
{
    __shared__ __align__(16) _Float16 smA[2][128 * 32];  // [oc_local][ic 32 halves]
    __shared__ __align__(16) _Float16 smB[2][128 * 32];  // [px_local][ic 32 halves]

    const int tid = threadIdx.x;
    const int b   = blockIdx.z;
    const int oc0 = blockIdx.y * 128;
    const int px0 = blockIdx.x * 128;

    const int lane = tid & 31;
    const int wid  = tid >> 5;
    const int woc  = wid >> 1;     // 0..3 -> oc sub-base 32*woc
    const int wpx  = wid & 1;      // 0..1 -> px sub-base 64*wpx
    const int lrow = lane & 15;
    const int hi   = lane >> 4;    // half-wave select

    v8f acc[2][4];
    #pragma unroll
    for (int f = 0; f < 2; ++f)
        #pragma unroll
        for (int g = 0; g < 4; ++g)
            #pragma unroll
            for (int e = 0; e < 8; ++e) acc[f][g][e] = 0.0f;

    auto stage = [&](int step, int buf) {
        int kk  = step >> 4;              // 0..8
        int ic0 = (step & 15) << 5;       // 0,32,..,480
        int dh  = kk / 3 - 1;
        int dw  = kk % 3 - 1;
        int row  = tid & 127;
        int hoff = (tid >> 7) << 4;       // 0 or 16 halves
        {   // A tile: Wp[b][kk][oc0+row][ic0 + hoff .. +hoff+15] -> LDS (async)
            unsigned lds = (unsigned)(unsigned long long)(uintptr_t)
                           (&smA[buf][row * 32 + hoff]);
            unsigned long long ga = (unsigned long long)(uintptr_t)(Wp
                + (((size_t)(b * KTAPS + kk) * OUT_CH + (oc0 + row)) << 9)
                + ic0 + hoff);
            asm volatile(
                "global_load_async_to_lds_b128 %0, %1, off\n\t"
                "global_load_async_to_lds_b128 %0, %1, off offset:16"
                :: "v"(lds), "v"(ga) : "memory");
        }
        {   // B tile: shifted Xt row (async), zero pad at borders (ds_store)
            int p = px0 + row;
            int h = (p >> 6) + dh;
            int w = (p & 63) + dw;
            _Float16* dstp = &smB[buf][row * 32 + hoff];
            if ((unsigned)h < (unsigned)HW && (unsigned)w < (unsigned)HW) {
                unsigned lds = (unsigned)(unsigned long long)(uintptr_t)dstp;
                unsigned long long ga = (unsigned long long)(uintptr_t)(Xt
                    + (((size_t)b * NPIX + (h << 6) + w) << 9)
                    + ic0 + hoff);
                asm volatile(
                    "global_load_async_to_lds_b128 %0, %1, off\n\t"
                    "global_load_async_to_lds_b128 %0, %1, off offset:16"
                    :: "v"(lds), "v"(ga) : "memory");
            } else {
                uint4 z = {0u, 0u, 0u, 0u};
                ((uint4*)dstp)[0] = z;
                ((uint4*)dstp)[1] = z;
            }
        }
    };

    stage(0, 0);

    for (int s = 0; s < 144; ++s) {
        // Wait for the previous stage's async copies, then rendezvous.
        asm volatile("s_wait_asynccnt 0x0" ::: "memory");
        __syncthreads();
        if (s + 1 < 144) stage(s + 1, (s + 1) & 1);

        const _Float16* A = smA[s & 1];
        const _Float16* B = smB[s & 1];

        // A fragments: lane holds row M=lrow; halves {0-7,16-23} (hi=0)
        // or {8-15,24-31} (hi=1), per ISA 16-bit A layout.
        Frag fa[2], fb[4];
        #pragma unroll
        for (int f = 0; f < 2; ++f) {
            const _Float16* p = A + (woc * 32 + f * 16 + lrow) * 32 + hi * 8;
            fa[f].h[0] = *(const v8h*)(p);
            fa[f].h[1] = *(const v8h*)(p + 16);
        }
        // B fragments: lane holds col N=lrow; halves {0-7,16-23} (hi=0)
        // or {8-15,24-31} (hi=1), matching the A layout.
        #pragma unroll
        for (int g = 0; g < 4; ++g) {
            const _Float16* p = B + (wpx * 64 + g * 16 + lrow) * 32 + hi * 8;
            fb[g].h[0] = *(const v8h*)(p);
            fb[g].h[1] = *(const v8h*)(p + 16);
        }

        #pragma unroll
        for (int f = 0; f < 2; ++f)
            #pragma unroll
            for (int g = 0; g < 4; ++g)
                acc[f][g] = __builtin_amdgcn_wmma_f32_16x16x32_f16(
                    false, fa[f].v, false, fb[g].v,
                    (short)0, acc[f][g], false, false);
    }

    // Epilogue: D layout — VGPR i: lanes 0-15 M=i, lanes 16-31 M=8+i; N=lane%16
    #pragma unroll
    for (int f = 0; f < 2; ++f) {
        int ocb = oc0 + woc * 32 + f * 16;
        #pragma unroll
        for (int g = 0; g < 4; ++g) {
            int p = px0 + wpx * 64 + g * 16 + lrow;
            float* op = out + (((size_t)b * OUT_CH + ocb) << 12) + p;
            #pragma unroll
            for (int i = 0; i < 8; ++i) {
                int m = i + hi * 8;
                op[(size_t)m << 12] = acc[f][g][i];
            }
        }
    }
}

// ---------------------------------------------------------------------------
extern "C" void kernel_launch(void* const* d_in, const int* in_sizes, int n_in,
                              void* d_out, int out_size, void* d_ws, size_t ws_size,
                              hipStream_t stream)
{
    const float* x          = (const float*)d_in[0];  // (8,512,64,64)
    const float* style      = (const float*)d_in[1];  // (8,512)
    const float* weight     = (const float*)d_in[2];  // (1,512,512,3,3)
    const float* mod_weight = (const float*)d_in[3];  // (512,512)
    const float* mod_bias   = (const float*)d_in[4];  // (512,)
    float* out = (float*)d_out;

    // Workspace carve-up (all 16B aligned):
    float*    s_mod = (float*)d_ws;                       // 4096 f32
    float*    demod = s_mod + BATCH * IN_CH;              // 4096 f32
    _Float16* Wp    = (_Float16*)(demod + BATCH * OUT_CH);          // 18,874,368 f16
    _Float16* Xt    = Wp + (size_t)BATCH * KTAPS * OUT_CH * IN_CH;  // 16,777,216 f16

    k_modstyle<<<dim3(16), dim3(256), 0, stream>>>(style, mod_weight, mod_bias, s_mod);
    k_demod<<<dim3(OUT_CH, BATCH), dim3(256), 0, stream>>>(weight, s_mod, demod);
    k_packw<<<dim3(73728), dim3(256), 0, stream>>>(weight, s_mod, demod, Wp);
    k_packx<<<dim3(NPIX / 32, IN_CH / 32, BATCH), dim3(32, 8), 0, stream>>>(x, Xt);
    k_mconv<<<dim3(NPIX / 128, OUT_CH / 128, BATCH), dim3(256), 0, stream>>>(Wp, Xt, out);
}
